// MiniCDDDInference_34110630265151
// MI455X (gfx1250) — matrix operand order repaired and checked by reading, not compiled, not run on hardware
//
#include <hip/hip_runtime.h>
#include <hip/hip_bf16.h>
#include <cstdint>

// Problem constants (from reference)
#define B_   256
#define T_   128
#define E_   32
#define S0_  512
#define S1_  1024
#define S2_  2048
#define LAT_ 512

typedef __attribute__((ext_vector_type(16))) __bf16 v16bf;
typedef __attribute__((ext_vector_type(8)))  __bf16 v8bf;
typedef __attribute__((ext_vector_type(8)))  float  v8f;

enum { EPI_GATES = 0, EPI_CAND = 1, EPI_DENSE = 2 };

union FragB16 { v16bf v; struct { v8bf lo, hi; } p; };

// WGP-scope prefetch (scope field = 0 -> pull into all cache levels incl. L0).
// offset:128 = 64 bf16 elements ahead (two 32-element K blocks); reuses the load's address register (no VALU adds).
__device__ __forceinline__ void prefetch_wgp_ahead(const void* p) {
  asm volatile("global_prefetch_b8 %0, off offset:128" :: "v"(p));
}

// ---- weight conversion: fp32 (K x N) -> bf16 transposed (N x K) ----
__global__ __launch_bounds__(256) void convert_transpose_kernel(
    const float* __restrict__ W, __bf16* __restrict__ Wt, int K, int N) {
  long idx = (long)blockIdx.x * blockDim.x + threadIdx.x;
  if (idx >= (long)K * N) return;
  int k = (int)(idx / N);
  int n = (int)(idx % N);
  Wt[(long)n * K + k] = (__bf16)W[idx];
}

// ---- straight fp32 -> bf16 convert (dense_w is already out x in) ----
__global__ __launch_bounds__(256) void convert_kernel(
    const float* __restrict__ W, __bf16* __restrict__ Wt, long total) {
  long idx = (long)blockIdx.x * blockDim.x + threadIdx.x;
  if (idx < total) Wt[idx] = (__bf16)W[idx];
}

// ---- embed ALL timesteps once: xall[b][t][e] (bf16) ----
__global__ __launch_bounds__(256) void embed_all_kernel(
    const int* __restrict__ seqs, const float* __restrict__ emb, __bf16* __restrict__ xall) {
  long idx = (long)blockIdx.x * blockDim.x + threadIdx.x;
  if (idx >= (long)B_ * T_ * E_) return;
  int e = (int)(idx % E_);
  long bt = idx / E_;                 // b*T_ + t
  int tok = seqs[bt];
  xall[idx] = (__bf16)emb[tok * E_ + e];
}

// ---- bf16 WMMA GEMM over a segmented A operand, with fused GRU epilogues ----
// A = [seg0 | seg1 | seg2] along K (each segment: base ptr, row stride, length; lengths
// are multiples of 32 so segment switches happen at uniform K-block boundaries).
// C(MxN) = A(MxK) * Wt(NxK)^T + bias, then epilogue.
// Block: 128 threads = 4 waves arranged 2(M) x 2(N); wave tile 32x32 (2x2 WMMA tiles);
// block tile 64(M) x 64(N); 2 b128 loads per WMMA via register reuse.
template <int EPI>
__global__ __launch_bounds__(128) void gemm_bf16_wmma(
    const __bf16* __restrict__ A0, int ld0, int len0,
    const __bf16* __restrict__ A1, int ld1, int len1,
    const __bf16* __restrict__ A2, int ld2, int len2,
    const __bf16* __restrict__ Wt, int K, int N,
    const float* __restrict__ bias,
    float*  __restrict__ hf, int H,       // fp32 recurrent state (B x H)
    __bf16* __restrict__ rh,              // GATES: write r*h (B x H)
    float*  __restrict__ zbuf,            // GATES: write z; CAND: read z
    __bf16* __restrict__ hbf,             // CAND: write bf16 masked h (B x H)
    __bf16* __restrict__ xnext, int ldxn, // CAND: next layer input (unmasked n)
    const int* __restrict__ lens, int t,
    float* __restrict__ outf)             // DENSE: final fp32 output
{
  const int lane = threadIdx.x & 31;
  const int wave = threadIdx.x >> 5;                   // 0..3
  const int m0 = blockIdx.y * 64 + (wave >> 1) * 32;   // wave's 32-row strip
  const int n0 = blockIdx.x * 64 + (wave & 1) * 32;    // wave's 32-col strip
  const int lm = lane & 15;   // m within tile (A), n within tile (B/D)
  const int lh = lane >> 4;   // half-wave selector

  v8f acc00 = {}, acc01 = {}, acc10 = {}, acc11 = {};
  int kw = 0;  // cumulative K offset into Wt rows
#pragma unroll
  for (int s = 0; s < 3; ++s) {
    const __bf16* P  = (s == 0) ? A0  : (s == 1) ? A1  : A2;
    const int     ld = (s == 0) ? ld0 : (s == 1) ? ld1 : ld2;
    const int     L  = (s == 0) ? len0 : (s == 1) ? len1 : len2;
    if (P == nullptr) break;  // uniform

    // A fragments: lane holds row m+lm; K chunks [lh*8..+7] and [16+lh*8..+7]
    const __bf16* aP0 = P + (long)(m0 + lm) * ld + lh * 8;
    const __bf16* aP1 = aP0 + (long)16 * ld;
    // B fragments: lane holds column n+lm (= row of Wt); K chunks [lh*8..+7] and [16+lh*8..+7]
    const __bf16* bP0 = Wt + (long)(n0 + lm) * K + kw + lh * 8;
    const __bf16* bP1 = bP0 + (long)16 * K;

#pragma unroll 2
    for (int k = 0; k < L; k += 32) {
      FragB16 a0, a1, b0, b1;
      a0.p.lo = *(const v8bf*)(aP0 + k);
      a0.p.hi = *(const v8bf*)(aP0 + k + 16);
      a1.p.lo = *(const v8bf*)(aP1 + k);
      a1.p.hi = *(const v8bf*)(aP1 + k + 16);
      b0.p.lo = *(const v8bf*)(bP0 + k);
      b0.p.hi = *(const v8bf*)(bP0 + k + 16);
      b1.p.lo = *(const v8bf*)(bP1 + k);
      b1.p.hi = *(const v8bf*)(bP1 + k + 16);
      if (k + 64 < L) {  // uniform; prefetch K block k+64 (two ahead) into near caches
        prefetch_wgp_ahead(aP0 + k);
        prefetch_wgp_ahead(aP1 + k);
        prefetch_wgp_ahead(bP0 + k);
        prefetch_wgp_ahead(bP1 + k);
      }
      acc00 = __builtin_amdgcn_wmma_f32_16x16x32_bf16(false, a0.v, false, b0.v, (short)0, acc00, false, false);
      acc01 = __builtin_amdgcn_wmma_f32_16x16x32_bf16(false, a0.v, false, b1.v, (short)0, acc01, false, false);
      acc10 = __builtin_amdgcn_wmma_f32_16x16x32_bf16(false, a1.v, false, b0.v, (short)0, acc10, false, false);
      acc11 = __builtin_amdgcn_wmma_f32_16x16x32_bf16(false, a1.v, false, b1.v, (short)0, acc11, false, false);
    }
    kw += L;
  }

  // D fragments: n = ntile + lm, m = mtile + lh*8 + v
  v8f accs[2][2] = { { acc00, acc01 }, { acc10, acc11 } };
#pragma unroll
  for (int ti = 0; ti < 2; ++ti) {
#pragma unroll
    for (int tj = 0; tj < 2; ++tj) {
      const int n = n0 + tj * 16 + lm;
      const float bn = bias[n];
#pragma unroll
      for (int v = 0; v < 8; ++v) {
        const int m = m0 + ti * 16 + lh * 8 + v;
        float s = accs[ti][tj][v] + bn;
        if (EPI == EPI_GATES) {
          float g = 1.0f / (1.0f + __expf(-s));           // sigmoid
          if (n < H)  rh[(long)m * H + n] = (__bf16)(g * hf[(long)m * H + n]);  // r*h
          else        zbuf[(long)m * H + (n - H)] = g;                          // z
        } else if (EPI == EPI_CAND) {
          float cand = tanhf(s);
          float hold = hf[(long)m * H + n];
          float z    = zbuf[(long)m * H + n];
          float nv   = z * hold + (1.0f - z) * cand;      // unmasked new value
          if (xnext) xnext[(long)m * ldxn + n] = (__bf16)nv;
          float hnew = (t < lens[m]) ? nv : hold;         // length-masked update
          hf[(long)m * H + n]  = hnew;
          hbf[(long)m * H + n] = (__bf16)hnew;
        } else { // EPI_DENSE
          outf[(long)m * N + n] = tanhf(s);
        }
      }
    }
  }
}

extern "C" void kernel_launch(void* const* d_in, const int* in_sizes, int n_in,
                              void* d_out, int out_size, void* d_ws, size_t ws_size,
                              hipStream_t stream) {
  const int*   seqs  = (const int*)  d_in[0];
  const int*   lens  = (const int*)  d_in[1];
  const float* emb   = (const float*)d_in[2];
  const float* g0_gk = (const float*)d_in[3];
  const float* g0_gb = (const float*)d_in[4];
  const float* g0_ck = (const float*)d_in[5];
  const float* g0_cb = (const float*)d_in[6];
  const float* g1_gk = (const float*)d_in[7];
  const float* g1_gb = (const float*)d_in[8];
  const float* g1_ck = (const float*)d_in[9];
  const float* g1_cb = (const float*)d_in[10];
  const float* g2_gk = (const float*)d_in[11];
  const float* g2_gb = (const float*)d_in[12];
  const float* g2_ck = (const float*)d_in[13];
  const float* g2_cb = (const float*)d_in[14];
  const float* dw    = (const float*)d_in[15];
  const float* db    = (const float*)d_in[16];
  float* out = (float*)d_out;

  char* ws = (char*)d_ws;
  size_t off = 0;
  auto take = [&](size_t bytes) -> void* {
    void* p = ws + off;
    off = (off + bytes + 255) & ~(size_t)255;
    return p;
  };

  // bf16 transposed weights (N x K)
  __bf16* wgT0 = (__bf16*)take((size_t)(2 * S0_) * (E_ + S0_) * 2);
  __bf16* wcT0 = (__bf16*)take((size_t)S0_ * (E_ + S0_) * 2);
  __bf16* wgT1 = (__bf16*)take((size_t)(2 * S1_) * (S0_ + S1_) * 2);
  __bf16* wcT1 = (__bf16*)take((size_t)S1_ * (S0_ + S1_) * 2);
  __bf16* wgT2 = (__bf16*)take((size_t)(2 * S2_) * (S1_ + S2_) * 2);
  __bf16* wcT2 = (__bf16*)take((size_t)S2_ * (S1_ + S2_) * 2);
  __bf16* dT   = (__bf16*)take((size_t)LAT_ * (S0_ + S1_ + S2_) * 2);
  // embedded inputs for all timesteps
  __bf16* xall = (__bf16*)take((size_t)B_ * T_ * E_ * 2);
  // fp32 recurrent states + bf16 shadows (A operands)
  float*  hf0  = (float*) take((size_t)B_ * S0_ * 4);
  float*  hf1  = (float*) take((size_t)B_ * S1_ * 4);
  float*  hf2  = (float*) take((size_t)B_ * S2_ * 4);
  __bf16* hb0  = (__bf16*)take((size_t)B_ * S0_ * 2);
  __bf16* hb1  = (__bf16*)take((size_t)B_ * S1_ * 2);
  __bf16* hb2  = (__bf16*)take((size_t)B_ * S2_ * 2);
  // r*h buffers (candidate A segment 1)
  __bf16* rh0  = (__bf16*)take((size_t)B_ * S0_ * 2);
  __bf16* rh1  = (__bf16*)take((size_t)B_ * S1_ * 2);
  __bf16* rh2  = (__bf16*)take((size_t)B_ * S2_ * 2);
  // unmasked layer outputs (next layer's x)
  __bf16* x1   = (__bf16*)take((size_t)B_ * S0_ * 2);
  __bf16* x2   = (__bf16*)take((size_t)B_ * S1_ * 2);
  float*  zb   = (float*) take((size_t)B_ * S2_ * 4);  // z gate scratch (max H)

  auto tr = [&](const float* W, __bf16* Wt, int K, int N) {
    long total = (long)K * N;
    convert_transpose_kernel<<<(unsigned)((total + 255) / 256), 256, 0, stream>>>(W, Wt, K, N);
  };
  tr(g0_gk, wgT0, E_ + S0_, 2 * S0_);
  tr(g0_ck, wcT0, E_ + S0_, S0_);
  tr(g1_gk, wgT1, S0_ + S1_, 2 * S1_);
  tr(g1_ck, wcT1, S0_ + S1_, S1_);
  tr(g2_gk, wgT2, S1_ + S2_, 2 * S2_);
  tr(g2_ck, wcT2, S1_ + S2_, S2_);
  {
    long total = (long)LAT_ * (S0_ + S1_ + S2_);
    convert_kernel<<<(unsigned)((total + 255) / 256), 256, 0, stream>>>(dw, dT, total);
  }
  {
    long total = (long)B_ * T_ * E_;
    embed_all_kernel<<<(unsigned)((total + 255) / 256), 256, 0, stream>>>(seqs, emb, xall);
  }

  // zero-init recurrent states (bf16 zero == 0x0000)
  hipMemsetAsync(hf0, 0, (size_t)B_ * S0_ * 4, stream);
  hipMemsetAsync(hf1, 0, (size_t)B_ * S1_ * 4, stream);
  hipMemsetAsync(hf2, 0, (size_t)B_ * S2_ * 4, stream);
  hipMemsetAsync(hb0, 0, (size_t)B_ * S0_ * 2, stream);
  hipMemsetAsync(hb1, 0, (size_t)B_ * S1_ * 2, stream);
  hipMemsetAsync(hb2, 0, (size_t)B_ * S2_ * 2, stream);

  const int LDX = T_ * E_;  // row stride of xall
  for (int t = 0; t < T_; ++t) {
    const __bf16* xt = xall + (long)t * E_;
    // ---- layer 0: I=32, H=512 ----
    gemm_bf16_wmma<EPI_GATES><<<dim3((2 * S0_) / 64, B_ / 64), 128, 0, stream>>>(
        xt, LDX, E_, hb0, S0_, S0_, nullptr, 0, 0,
        wgT0, E_ + S0_, 2 * S0_, g0_gb,
        hf0, S0_, rh0, zb, nullptr, nullptr, 0, nullptr, 0, nullptr);
    gemm_bf16_wmma<EPI_CAND><<<dim3(S0_ / 64, B_ / 64), 128, 0, stream>>>(
        xt, LDX, E_, rh0, S0_, S0_, nullptr, 0, 0,
        wcT0, E_ + S0_, S0_, g0_cb,
        hf0, S0_, nullptr, zb, hb0, x1, S0_, lens, t, nullptr);
    // ---- layer 1: I=512, H=1024 ----
    gemm_bf16_wmma<EPI_GATES><<<dim3((2 * S1_) / 64, B_ / 64), 128, 0, stream>>>(
        x1, S0_, S0_, hb1, S1_, S1_, nullptr, 0, 0,
        wgT1, S0_ + S1_, 2 * S1_, g1_gb,
        hf1, S1_, rh1, zb, nullptr, nullptr, 0, nullptr, 0, nullptr);
    gemm_bf16_wmma<EPI_CAND><<<dim3(S1_ / 64, B_ / 64), 128, 0, stream>>>(
        x1, S0_, S0_, rh1, S1_, S1_, nullptr, 0, 0,
        wcT1, S0_ + S1_, S1_, g1_cb,
        hf1, S1_, nullptr, zb, hb1, x2, S1_, lens, t, nullptr);
    // ---- layer 2: I=1024, H=2048 ----
    gemm_bf16_wmma<EPI_GATES><<<dim3((2 * S2_) / 64, B_ / 64), 128, 0, stream>>>(
        x2, S1_, S1_, hb2, S2_, S2_, nullptr, 0, 0,
        wgT2, S1_ + S2_, 2 * S2_, g2_gb,
        hf2, S2_, rh2, zb, nullptr, nullptr, 0, nullptr, 0, nullptr);
    gemm_bf16_wmma<EPI_CAND><<<dim3(S2_ / 64, B_ / 64), 128, 0, stream>>>(
        x2, S1_, S1_, rh2, S2_, S2_, nullptr, 0, 0,
        wcT2, S1_ + S2_, S2_, g2_cb,
        hf2, S2_, nullptr, zb, hb2, nullptr, 0, lens, t, nullptr);
  }

  // ---- final dense + tanh: A = [hb0 | hb1 | hb2] as 3 segments ----
  gemm_bf16_wmma<EPI_DENSE><<<dim3(LAT_ / 64, B_ / 64), 128, 0, stream>>>(
      hb0, S0_, S0_, hb1, S1_, S1_, hb2, S2_, S2_,
      dT, S0_ + S1_ + S2_, LAT_, db,
      nullptr, 0, nullptr, nullptr, nullptr, nullptr, 0, nullptr, 0, out);
}
